// MatGPT_64364379898248
// MI455X (gfx1250) — matrix operand order repaired and checked by reading, not compiled, not run on hardware
//
#include <hip/hip_runtime.h>
#include <math.h>

typedef _Float16 f16;
typedef __attribute__((ext_vector_type(16))) _Float16 v16h;
typedef __attribute__((ext_vector_type(8)))  float    v8f;

__device__ __forceinline__ v8f wmma16(v16h a, v16h b, v8f c) {
    return __builtin_amdgcn_wmma_f32_16x16x32_f16(false, a, false, b, (short)0, c, false, false);
}

__device__ __forceinline__ float gelu_exact(float v) {
    return 0.5f * v * (1.0f + erff(v * 0.70710678118654752f));
}

union AFrag { float4 f4[2]; v16h v; };
union Pack2 { f16 h[2]; unsigned u; };
union H8    { float4 f4; f16 h[8]; };

// ---------------- embedding: x = tok_emb[idx] + pos_emb ----------------
__global__ __launch_bounds__(256) void embed_kernel(
    const int* __restrict__ idx, const float* __restrict__ tok,
    const float* __restrict__ pos, float* __restrict__ x) {
    int gid = blockIdx.x * 256 + threadIdx.x;      // over M*D = 2048*1024
    int c = gid & 1023;
    int r = gid >> 10;
    int t = r & 1023;
    x[gid] = tok[(size_t)idx[r] * 1024 + c] + pos[(size_t)t * 1024 + c];
}

// ---------------- layernorm over D=1024, output f16 ----------------
__global__ __launch_bounds__(256) void ln_kernel(
    const float* __restrict__ x, const float* __restrict__ g,
    const float* __restrict__ b, f16* __restrict__ out) {
    __shared__ float redA[8], redB[8];
    int row = blockIdx.x, tid = threadIdx.x;
    int wv = tid >> 5, lane = tid & 31;
    const float* xr = x + (size_t)row * 1024;
    float vals[4];
    float s = 0.f, s2 = 0.f;
#pragma unroll
    for (int j = 0; j < 4; ++j) {
        float v = xr[tid + j * 256];
        vals[j] = v; s += v; s2 += v * v;
    }
#pragma unroll
    for (int msk = 16; msk >= 1; msk >>= 1) {
        s  += __shfl_xor(s,  msk, 32);
        s2 += __shfl_xor(s2, msk, 32);
    }
    if (lane == 0) { redA[wv] = s; redB[wv] = s2; }
    __syncthreads();
    float tot = 0.f, tot2 = 0.f;
#pragma unroll
    for (int j = 0; j < 8; ++j) { tot += redA[j]; tot2 += redB[j]; }
    float mean = tot * (1.0f / 1024.0f);
    float var  = tot2 * (1.0f / 1024.0f) - mean * mean;
    float rstd = rsqrtf(var + 1e-5f);
    f16* outr = out + (size_t)row * 1024;
#pragma unroll
    for (int j = 0; j < 4; ++j) {
        int c = tid + j * 256;
        outr[c] = (f16)((vals[j] - mean) * rstd * g[c] + b[c]);
    }
}

// ---------------- RoPE in place on f16 (B*T,1024), heads of 64, first 32 rotated ----------------
__global__ __launch_bounds__(256) void rope_kernel(f16* __restrict__ u) {
    int gid = blockIdx.x * 256 + threadIdx.x;  // 2048 * 16 * 16 threads
    int j = gid & 15;
    int h = (gid >> 4) & 15;
    int r = gid >> 8;
    int t = r & 1023;
    float inv0 = powf(10000.0f, -(float)j * (1.0f / 32.0f));
    float inv1 = powf(10000.0f, -(float)(j + 16) * (1.0f / 32.0f));
    float a0 = (float)t * inv0, a1 = (float)t * inv1;
    size_t base = (size_t)r * 1024 + h * 64;
    float u0 = (float)u[base + j], u1 = (float)u[base + j + 16];
    u[base + j]      = (f16)(u0 * cosf(a0) - u1 * sinf(a0));
    u[base + j + 16] = (f16)(u1 * cosf(a1) + u0 * sinf(a1));
}

// ---------------- WMMA GEMM: C[M,N] = A_f16[M,K] @ W_f32[K,N], templated epilogue ----------------
// block tile 128x128, 8 waves (4M x 2N); wave computes 32(M) x 64(N) = 8 WMMA tiles / K-step,
// reusing 4 B fragments across 2 A fragments (2x WMMA density per LDS byte).
// Double-buffered LDS for W (fragment-ordered), software-pipelined W+A prefetch.
template<bool BIAS, bool GELU, bool MUL, bool RESID, bool OUTF>
__global__ __launch_bounds__(256) void gemm_wmma(
    const f16* __restrict__ A, const float* __restrict__ W,
    const float* __restrict__ bias, const float* __restrict__ resid,
    const f16* __restrict__ mul, float* __restrict__ outF, f16* __restrict__ outH,
    int M, int N, int K) {
    __shared__ __align__(32) f16 sB[2][32 * 128];   // fragment-ordered, double buffered
    int tid = threadIdx.x;
    int wv = tid >> 5, lane = tid & 31;
    int half = lane >> 4, lrow = lane & 15;
    int mo = (wv & 3) * 32;
    int no = (wv >> 2) * 64;
    int m0 = blockIdx.y * 128, n0 = blockIdx.x * 128;

    // staging role: kp -> k rows {2kp, 2kp+1}, cg -> cols [cg*8, cg*8+8)
    int kp = tid >> 4, cg = tid & 15;
    int laneBase = (cg >> 1) * 32 + (((kp >> 3) << 4) | ((cg & 1) * 8));
    int eoff     = (kp * 2) & 15;
    const float* wSrc = W + (size_t)(kp * 2) * N + n0 + cg * 8;

    auto ldW = [&](int k0, float4 r[4]) {
        const float* w0 = wSrc + (size_t)k0 * N;
        r[0] = *(const float4*)w0;       r[1] = *(const float4*)(w0 + 4);
        r[2] = *(const float4*)(w0 + N); r[3] = *(const float4*)(w0 + N + 4);
    };
    auto stW = [&](const float4 r[4], int buf) {
        float va[8] = {r[0].x, r[0].y, r[0].z, r[0].w, r[1].x, r[1].y, r[1].z, r[1].w};
        float vb[8] = {r[2].x, r[2].y, r[2].z, r[2].w, r[3].x, r[3].y, r[3].z, r[3].w};
#pragma unroll
        for (int j = 0; j < 8; ++j) {
            Pack2 p; p.h[0] = (f16)va[j]; p.h[1] = (f16)vb[j];
            *(unsigned*)(sB[buf] + ((laneBase + j) * 16 + eoff)) = p.u;
        }
    };
    const f16* aBase0 = A + (size_t)(m0 + mo + lrow) * K;
    const f16* aBase1 = aBase0 + (size_t)16 * K;
    auto ldA = [&](const f16* ab, int k0) {
        AFrag au;
        au.f4[0] = *(const float4*)(ab + k0 + (half << 4));
        au.f4[1] = *(const float4*)(ab + k0 + (half << 4) + 8);
        return au.v;
    };

    v8f zero = {0.f,0.f,0.f,0.f,0.f,0.f,0.f,0.f};
    v8f acc[2][4] = {{zero, zero, zero, zero}, {zero, zero, zero, zero}};

    // prologue: stage first W tile, load first A fragments
    { float4 r[4]; ldW(0, r); stW(r, 0); }
    v16h aCur0 = ldA(aBase0, 0), aCur1 = ldA(aBase1, 0);
    __syncthreads();

    int nsteps = K >> 5, cur = 0;
    for (int s = 0; s < nsteps - 1; ++s) {
        int k0 = s << 5;
        float4 rn[4]; ldW(k0 + 32, rn);            // prefetch next W (global)
        v16h aN0 = ldA(aBase0, k0 + 32);           // prefetch next A (global)
        v16h aN1 = ldA(aBase1, k0 + 32);
        v16h bf[4];
#pragma unroll
        for (int tn = 0; tn < 4; ++tn)
            bf[tn] = *(const v16h*)(sB[cur] + (((no >> 4) + tn) * 32 + lane) * 16);
#pragma unroll
        for (int tn = 0; tn < 4; ++tn) {
            acc[0][tn] = wmma16(aCur0, bf[tn], acc[0][tn]);
            acc[1][tn] = wmma16(aCur1, bf[tn], acc[1][tn]);
        }
        stW(rn, cur ^ 1);                          // fill other buffer
        __syncthreads();
        cur ^= 1;
        aCur0 = aN0; aCur1 = aN1;
    }
    {   // peeled last step
        v16h bf[4];
#pragma unroll
        for (int tn = 0; tn < 4; ++tn)
            bf[tn] = *(const v16h*)(sB[cur] + (((no >> 4) + tn) * 32 + lane) * 16);
#pragma unroll
        for (int tn = 0; tn < 4; ++tn) {
            acc[0][tn] = wmma16(aCur0, bf[tn], acc[0][tn]);
            acc[1][tn] = wmma16(aCur1, bf[tn], acc[1][tn]);
        }
    }

    // epilogue: C layout -> row = sub*16 + i + 8*half, col = lrow
#pragma unroll
    for (int sub = 0; sub < 2; ++sub) {
#pragma unroll
        for (int tn = 0; tn < 4; ++tn) {
#pragma unroll
            for (int i = 0; i < 8; ++i) {
                int row = m0 + mo + sub * 16 + i + 8 * half;
                int col = n0 + no + tn * 16 + lrow;
                float v = acc[sub][tn][i];
                if constexpr (BIAS)  v += bias[col];
                if constexpr (GELU)  v = gelu_exact(v);
                if constexpr (MUL)   v *= (float)mul[(size_t)row * N + col];
                if constexpr (RESID) v += resid[(size_t)row * N + col];
                if constexpr (OUTF)  outF[(size_t)row * N + col] = v;
                else                 outH[(size_t)row * N + col] = (f16)v;
            }
        }
    }
}

// ---------------- flash-attention with WMMA ----------------
// grid (qb=16, h=16, b=2), block 128 (4 waves), each wave owns 16 q rows.
__global__ __launch_bounds__(128) void attn_wmma(
    const f16* __restrict__ q, const f16* __restrict__ k, const f16* __restrict__ v,
    const float* __restrict__ relb, f16* __restrict__ y) {
    __shared__ __align__(32) f16 sK[32 * 64];    // row-major [kcol][d]
    __shared__ __align__(32) f16 sV[32 * 64];    // fragment-ordered
    __shared__ __align__(32) f16 sP[4][512];     // fragment-ordered per wave
    int tid = threadIdx.x;
    int wv = tid >> 5, lane = tid & 31;
    int half = lane >> 4, lrow = lane & 15;
    int qb = blockIdx.x, h = blockIdx.y, b = blockIdx.z;
    int qbase = qb * 64;

    // Q A-fragments: two contiguous 16B runs per lane, straight from global
    v16h qa[2];
    {
        const f16* qp = q + ((size_t)(b * 1024 + qbase + wv * 16 + lrow)) * 1024 + h * 64;
#pragma unroll
        for (int dc = 0; dc < 2; ++dc) {
            AFrag au;
            au.f4[0] = *(const float4*)(qp + dc * 32 + (half << 4));
            au.f4[1] = *(const float4*)(qp + dc * 32 + (half << 4) + 8);
            qa[dc] = au.v;
        }
    }

    v8f zero = {0.f,0.f,0.f,0.f,0.f,0.f,0.f,0.f};
    v8f o[4] = {zero, zero, zero, zero};
    float mrun[8], lrun[8];
#pragma unroll
    for (int i = 0; i < 8; ++i) { mrun[i] = -1e30f; lrun[i] = 0.f; }

    // staging roles
    int krow = tid >> 2, kc16 = (tid & 3) * 16;          // K copy: 32B per thread
    int kpv = tid >> 3, cgv = tid & 7;                   // V permute
    int laneBaseV = (cgv >> 1) * 32 + (((kpv >> 3) << 4) | ((cgv & 1) * 8));
    int eoffV    = (kpv * 2) & 15;

    int nk = (qb + 1) * 2;  // 32-wide k blocks under causal limit
    for (int kb = 0; kb < nk; ++kb) {
        int kbase = kb * 32;
        {   // K tile: plain row-major copy (frag reads are contiguous later)
            const f16* ks = k + (size_t)(b * 1024 + kbase + krow) * 1024 + h * 64 + kc16;
            f16* kd = sK + krow * 64 + kc16;
            *(float4*)kd       = *(const float4*)ks;
            *(float4*)(kd + 8) = *(const float4*)(ks + 8);
            // V tile: permute to fragment order
            const f16* v0 = v + (size_t)(b * 1024 + kbase + kpv * 2) * 1024 + h * 64 + cgv * 8;
            H8 h0, h1;
            h0.f4 = *(const float4*)v0;
            h1.f4 = *(const float4*)(v0 + 1024);
#pragma unroll
            for (int j = 0; j < 8; ++j) {
                Pack2 p; p.h[0] = h0.h[j]; p.h[1] = h1.h[j];
                *(unsigned*)(sV + ((laneBaseV + j) * 16 + eoffV)) = p.u;
            }
        }
        __syncthreads();

        // S = Q @ K^T : load all 4 K fragments, then 4 WMMAs back-to-back
        v16h kf[4];
#pragma unroll
        for (int dc = 0; dc < 2; ++dc) {
            kf[dc * 2]     = *(const v16h*)(sK + lrow * 64 + dc * 32 + (half << 4));
            kf[dc * 2 + 1] = *(const v16h*)(sK + (16 + lrow) * 64 + dc * 32 + (half << 4));
        }
        v8f s0 = zero, s1 = zero;
        s0 = wmma16(qa[0], kf[0], s0);
        s1 = wmma16(qa[0], kf[1], s1);
        s0 = wmma16(qa[1], kf[2], s0);
        s1 = wmma16(qa[1], kf[3], s1);

        // online softmax update; write P swizzled to fragment order
        int pLane = 16;   // k-cols 16..31 go to lane-half 1, element = k-col % 16
        int pE    = lrow;
        float fac[8];
#pragma unroll
        for (int i = 0; i < 8; ++i) {
            int tq = qbase + wv * 16 + i + 8 * half;
            int tk0 = kbase + lrow, tk1 = kbase + 16 + lrow;
            float e0 = s0[i] * 0.125f + relb[(size_t)(tq - tk0 + 1023) * 16 + h];
            float e1 = s1[i] * 0.125f + relb[(size_t)(tq - tk1 + 1023) * 16 + h];
            if (tk0 > tq) e0 = -1e30f;
            if (tk1 > tq) e1 = -1e30f;
            float ml = fmaxf(e0, e1);
            for (int msk = 8; msk >= 1; msk >>= 1)
                ml = fmaxf(ml, __shfl_xor(ml, msk, 32));
            float mnew = fmaxf(mrun[i], ml);
            float p0 = __expf(e0 - mnew), p1 = __expf(e1 - mnew);
            float ps = p0 + p1;
            for (int msk = 8; msk >= 1; msk >>= 1)
                ps += __shfl_xor(ps, msk, 32);
            float f = __expf(mrun[i] - mnew);
            lrun[i] = lrun[i] * f + ps;
            mrun[i] = mnew;
            fac[i] = f;
            int row = i + 8 * half;
            sP[wv][row * 16 + pE]           = (f16)p0;
            sP[wv][(pLane + row) * 16 + pE] = (f16)p1;
        }
#pragma unroll
        for (int t = 0; t < 4; ++t)
#pragma unroll
            for (int i = 0; i < 8; ++i) o[t][i] *= fac[i];
        __syncthreads();

        // P @ V : load all fragments, then 4 WMMAs back-to-back
        v16h pa = *(const v16h*)(&sP[wv][lane * 16]);
        v16h vf[4];
#pragma unroll
        for (int t = 0; t < 4; ++t)
            vf[t] = *(const v16h*)(sV + (t * 32 + lane) * 16);
#pragma unroll
        for (int t = 0; t < 4; ++t)
            o[t] = wmma16(pa, vf[t], o[t]);
        __syncthreads();
    }

    // normalize and store y (f16, (B*T, D))
#pragma unroll
    for (int t = 0; t < 4; ++t)
#pragma unroll
        for (int i = 0; i < 8; ++i) {
            int tq = qbase + wv * 16 + i + 8 * half;
            float val = o[t][i] / lrun[i];
            y[((size_t)(b * 1024 + tq)) * 1024 + h * 64 + t * 16 + lrow] = (f16)val;
        }
}

extern "C" void kernel_launch(void* const* d_in, const int* in_sizes, int n_in,
                              void* d_out, int out_size, void* d_ws, size_t ws_size,
                              hipStream_t stream) {
    (void)in_sizes; (void)n_in; (void)out_size; (void)ws_size;
    const int B = 2, T = 1024, D = 1024, H = 16, L = 8, V = 32000, F = 4096;
    const int M = B * T;

    const int*   idx  = (const int*)  d_in[0];
    const float* tok  = (const float*)d_in[1];
    const float* pos  = (const float*)d_in[2];
    const float* ln1g = (const float*)d_in[3];
    const float* ln1b = (const float*)d_in[4];
    const float* wq   = (const float*)d_in[5];
    const float* wk   = (const float*)d_in[6];
    const float* wvw  = (const float*)d_in[7];
    const float* wo   = (const float*)d_in[8];
    const float* bo   = (const float*)d_in[9];
    const float* relb = (const float*)d_in[10];
    const float* ln2g = (const float*)d_in[11];
    const float* ln2b = (const float*)d_in[12];
    const float* w1   = (const float*)d_in[13];
    const float* b1   = (const float*)d_in[14];
    const float* w2   = (const float*)d_in[15];
    const float* b2   = (const float*)d_in[16];
    const float* w3   = (const float*)d_in[17];
    const float* b3   = (const float*)d_in[18];
    const float* lnfg = (const float*)d_in[19];
    const float* lnfb = (const float*)d_in[20];
    const float* head = (const float*)d_in[21];
    float* out = (float*)d_out;

    size_t off = 0;
    auto alloc = [&](size_t bytes) {
        void* p = (char*)d_ws + off;
        off += (bytes + 255) & ~(size_t)255;
        return p;
    };
    float* x   = (float*)alloc((size_t)M * D * 4);
    f16* hbf   = (f16*)alloc((size_t)M * D * 2);
    f16* qbf   = (f16*)alloc((size_t)M * D * 2);
    f16* kbf   = (f16*)alloc((size_t)M * D * 2);
    f16* vbf   = (f16*)alloc((size_t)M * D * 2);
    f16* ybf   = (f16*)alloc((size_t)M * D * 2);
    f16* g1    = (f16*)alloc((size_t)M * F * 2);
    f16* ubf   = (f16*)alloc((size_t)M * F * 2);

    embed_kernel<<<M * D / 256, 256, 0, stream>>>(idx, tok, pos, x);

    for (int l = 0; l < L; ++l) {
        ln_kernel<<<M, 256, 0, stream>>>(x, ln1g + l * D, ln1b + l * D, hbf);
        gemm_wmma<false,false,false,false,false><<<dim3(D / 128, M / 128), 256, 0, stream>>>(
            hbf, wq + (size_t)l * D * D, nullptr, nullptr, nullptr, nullptr, qbf, M, D, D);
        gemm_wmma<false,false,false,false,false><<<dim3(D / 128, M / 128), 256, 0, stream>>>(
            hbf, wk + (size_t)l * D * D, nullptr, nullptr, nullptr, nullptr, kbf, M, D, D);
        gemm_wmma<false,false,false,false,false><<<dim3(D / 128, M / 128), 256, 0, stream>>>(
            hbf, wvw + (size_t)l * D * D, nullptr, nullptr, nullptr, nullptr, vbf, M, D, D);
        rope_kernel<<<M * H * 16 / 256, 256, 0, stream>>>(qbf);
        rope_kernel<<<M * H * 16 / 256, 256, 0, stream>>>(kbf);
        attn_wmma<<<dim3(T / 64, H, B), 128, 0, stream>>>(
            qbf, kbf, vbf, relb + (size_t)l * (2 * T - 1) * H, ybf);
        gemm_wmma<true,false,false,true,true><<<dim3(D / 128, M / 128), 256, 0, stream>>>(
            ybf, wo + (size_t)l * D * D, bo + l * D, x, nullptr, x, nullptr, M, D, D);
        ln_kernel<<<M, 256, 0, stream>>>(x, ln2g + l * D, ln2b + l * D, hbf);
        gemm_wmma<true,true,false,false,false><<<dim3(F / 128, M / 128), 256, 0, stream>>>(
            hbf, w1 + (size_t)l * D * F, b1 + l * F, nullptr, nullptr, nullptr, g1, M, F, D);
        gemm_wmma<true,true,true,false,false><<<dim3(F / 128, M / 128), 256, 0, stream>>>(
            hbf, w2 + (size_t)l * D * F, b2 + l * F, nullptr, g1, nullptr, ubf, M, F, D);
        gemm_wmma<true,false,false,true,true><<<dim3(D / 128, M / 128), 256, 0, stream>>>(
            ubf, w3 + (size_t)l * F * D, b3 + l * D, x, nullptr, x, nullptr, M, D, F);
    }

    ln_kernel<<<M, 256, 0, stream>>>(x, lnfg, lnfb, hbf);
    gemm_wmma<false,false,false,false,true><<<dim3(V / 128, M / 128), 256, 0, stream>>>(
        hbf, head, nullptr, nullptr, nullptr, out, nullptr, M, V, D);
}
